// GCN_70360154243503
// MI455X (gfx1250) — matrix operand order repaired and checked by reading, not compiled, not run on hardware
//
#include <hip/hip_runtime.h>

#define NODES 100000
#define EDGES 1600000

typedef __attribute__((ext_vector_type(16))) _Float16 v16h;
typedef __attribute__((ext_vector_type(8)))  float    v8f;

// ---------------------------------------------------------------------------
// Dense GEMM  Y[M,N] = X[M,K] @ W[K,N]  via v_wmma_f32_16x16x32_f16.
// One wave32 per 16x16 output tile; K in {16,32} (zero-padded to 32).
// W is staged once per block into LDS, transposed + f16-converted, so each
// lane's B fragment is two 16B chunks at K = hi*8 and 16+hi*8 (2x ds_load_b128).
// A fragment loads are float4-vectorized (rows of X are contiguous).
// Fragment layouts per CDNA5 ISA 7.12.2:
//   A (16x32 f16): lane = hi*16+l16, row = tm*16+l16,
//     half j=0..7 -> K = hi*8+j ; half j=8..15 -> K = 16+hi*8+(j-8)
//   B (32x16 f16): col = tn*16+l16, same K mapping as A
//   C/D (16x16 f32): VGPR r -> row tm*16+hi*8+r, col tn*16+l16
// ---------------------------------------------------------------------------
template<int K, int N>
__global__ void gcn_gemm_wmma(const float* __restrict__ X,
                              const float* __restrict__ W,
                              float* __restrict__ Y, int M) {
  static_assert((K == 16 || K == 32) && (N % 16) == 0, "shape");

  // ---- stage W into LDS, transposed [n][k], f16, K zero-padded to 32 ----
  __shared__ _Float16 sWT[N * 32];
  for (int t = threadIdx.x; t < N * 32; t += blockDim.x) {
    const int n = t >> 5;        // column of W
    const int k = t & 31;        // K index
    sWT[t] = (k < K) ? (_Float16)W[k * N + n] : (_Float16)0.f;
  }
  __syncthreads();

  const int lane = threadIdx.x & 31;
  const int wave = (blockIdx.x * blockDim.x + threadIdx.x) >> 5;
  const int ntn  = N / 16;
  const int tm   = wave / ntn;
  const int tn   = wave % ntn;
  if (tm * 16 >= M) return;                     // wave-uniform exit
  const int hi  = lane >> 4;                    // lane half: 0 or 1
  const int l16 = lane & 15;

  // ---- A fragment: vectorized float4 loads from row tm*16+l16 ----
  const int     rowm = tm * 16 + l16;
  const float4* xv   = (const float4*)(X + (long)rowm * K);
  const int     kbA  = hi * 8;                  // 0 or 8
  v16h a;
  {
    const float4 p0 = xv[(kbA >> 2) + 0];       // K = kbA..kbA+3
    const float4 p1 = xv[(kbA >> 2) + 1];       // K = kbA+4..kbA+7
    a[0] = (_Float16)p0.x; a[1] = (_Float16)p0.y;
    a[2] = (_Float16)p0.z; a[3] = (_Float16)p0.w;
    a[4] = (_Float16)p1.x; a[5] = (_Float16)p1.y;
    a[6] = (_Float16)p1.z; a[7] = (_Float16)p1.w;
    if constexpr (K == 32) {
      const float4 q0 = xv[((16 + kbA) >> 2) + 0];
      const float4 q1 = xv[((16 + kbA) >> 2) + 1];
      a[ 8] = (_Float16)q0.x; a[ 9] = (_Float16)q0.y;
      a[10] = (_Float16)q0.z; a[11] = (_Float16)q0.w;
      a[12] = (_Float16)q1.x; a[13] = (_Float16)q1.y;
      a[14] = (_Float16)q1.z; a[15] = (_Float16)q1.w;
    } else {
#pragma unroll
      for (int j = 8; j < 16; ++j) a[j] = (_Float16)0.f;
    }
  }

  // ---- B fragment: two 16B LDS reads per lane (K = kbB.., 16+kbB..) ----
  const int coln = tn * 16 + l16;
  const int kbB  = hi * 8;                      // 0 or 8
  v16h b;
#pragma unroll
  for (int j = 0; j < 8; ++j) {
    b[j]     = sWT[coln * 32 + kbB + j];
    b[8 + j] = sWT[coln * 32 + 16 + kbB + j];
  }

  v8f c = {};
  c = __builtin_amdgcn_wmma_f32_16x16x32_f16(false, a, false, b,
                                             (short)0, c, false, false);
#pragma unroll
  for (int r = 0; r < 8; ++r) {
    const int m = tm * 16 + hi * 8 + r;
    Y[(long)m * N + coln] = c[r];
  }
}

// ---------------------------------------------------------------------------
// Degree / normalization helpers
// ---------------------------------------------------------------------------
__global__ void gcn_deg_init(float* __restrict__ deg, int n) {
  const int i = blockIdx.x * blockDim.x + threadIdx.x;
  if (i < n) deg[i] = 1.0f;                     // self-loop
}

__global__ void gcn_deg_count(const int* __restrict__ col,
                              float* __restrict__ deg, int e) {
  const int i = blockIdx.x * blockDim.x + threadIdx.x;
  if (i < e) unsafeAtomicAdd(&deg[col[i]], 1.0f);
}

__global__ void gcn_dinv(float* __restrict__ deg, int n) {
  const int i = blockIdx.x * blockDim.x + threadIdx.x;
  if (i < n) {
    const float d = deg[i];
    deg[i] = (d > 0.f) ? rsqrtf(d) : 0.f;       // in-place deg -> dinv
  }
}

// agg[n,f] = dinv[n]^2 * h[n,f]   (self-loop term; initializes agg)
template<int F>
__global__ void gcn_selfloop(const float* __restrict__ h,
                             const float* __restrict__ dinv,
                             float* __restrict__ agg, int total) {
  const int t = blockIdx.x * blockDim.x + threadIdx.x;
  if (t >= total) return;
  const float di = dinv[t / F];
  agg[t] = di * di * h[t];
}

// agg[col,f] += dinv[row]*dinv[col]*h[row,f]   one thread per (edge, feature)
template<int F>
__global__ void gcn_edge_scatter(const int* __restrict__ row,
                                 const int* __restrict__ col,
                                 const float* __restrict__ dinv,
                                 const float* __restrict__ h,
                                 float* __restrict__ agg, int e) {
  const long t  = (long)blockIdx.x * blockDim.x + threadIdx.x;
  const int  ei = (int)(t / F);
  const int  f  = (int)(t % F);
  if (ei >= e) return;
  const int r = row[ei];
  const int c = col[ei];
  const float s = dinv[r] * dinv[c];
  unsafeAtomicAdd(&agg[(long)c * F + f], s * h[(long)r * F + f]);
}

template<int F, bool RELU>
__global__ void gcn_bias_act(float* __restrict__ h,
                             const float* __restrict__ bias, int total) {
  const int t = blockIdx.x * blockDim.x + threadIdx.x;
  if (t >= total) return;
  const float v = h[t] + bias[t % F];
  h[t] = RELU ? fmaxf(v, 0.f) : v;
}

// ---------------------------------------------------------------------------
extern "C" void kernel_launch(void* const* d_in, const int* in_sizes, int n_in,
                              void* d_out, int out_size, void* d_ws, size_t ws_size,
                              hipStream_t stream) {
  const float* x  = (const float*)d_in[0];      // [N,16]
  const int*   ei = (const int*)  d_in[1];      // [2,E]
  const float* W1 = (const float*)d_in[2];      // [16,32]
  const float* b1 = (const float*)d_in[3];      // [32]
  const float* W2 = (const float*)d_in[4];      // [32,16]
  const float* b2 = (const float*)d_in[5];      // [16]
  const int* row = ei;
  const int* col = ei + EDGES;
  float* out = (float*)d_out;                   // [N,16]

  float* ws   = (float*)d_ws;
  float* dinv = ws;                               // NODES floats
  float* h1   = dinv + NODES;                     // NODES*32
  float* agg1 = h1   + (size_t)NODES * 32;        // NODES*32
  float* h2   = agg1 + (size_t)NODES * 32;        // NODES*16

  const int B = 256;
  const int gN   = (NODES + B - 1) / B;
  const int gE   = (EDGES + B - 1) / B;
  const int g32  = (NODES * 32 + B - 1) / B;
  const int g16  = (NODES * 16 + B - 1) / B;
  const int gE32 = (int)(((long)EDGES * 32 + B - 1) / B);
  const int gE16 = (int)(((long)EDGES * 16 + B - 1) / B);

  // normalization (shared by both layers)
  gcn_deg_init <<<gN, B, 0, stream>>>(dinv, NODES);
  gcn_deg_count<<<gE, B, 0, stream>>>(col, dinv, EDGES);
  gcn_dinv     <<<gN, B, 0, stream>>>(dinv, NODES);

  // ---- layer 1: h1 = x @ W1 ; agg1 = norm-aggregate(h1) ; relu(+b1) ----
  {
    const int waves  = (NODES / 16) * (32 / 16);        // 12500
    const int blocks = (waves * 32 + B - 1) / B;
    gcn_gemm_wmma<16, 32><<<blocks, B, 0, stream>>>(x, W1, h1, NODES);
  }
  gcn_selfloop<32>      <<<g32,  B, 0, stream>>>(h1, dinv, agg1, NODES * 32);
  gcn_edge_scatter<32>  <<<gE32, B, 0, stream>>>(row, col, dinv, h1, agg1, EDGES);
  gcn_bias_act<32, true><<<g32,  B, 0, stream>>>(agg1, b1, NODES * 32);

  // ---- layer 2: h2 = agg1 @ W2 ; out = norm-aggregate(h2) + b2 ----
  {
    const int waves  = (NODES / 16) * (16 / 16);        // 6250
    const int blocks = (waves * 32 + B - 1) / B;
    gcn_gemm_wmma<32, 16><<<blocks, B, 0, stream>>>(agg1, W2, h2, NODES);
  }
  gcn_selfloop<16>       <<<g16,  B, 0, stream>>>(h2, dinv, out, NODES * 16);
  gcn_edge_scatter<16>   <<<gE16, B, 0, stream>>>(row, col, dinv, h2, out, EDGES);
  gcn_bias_act<16, false><<<g16,  B, 0, stream>>>(out, b2, NODES * 16);
}
